// S_SSD_41532333752670
// MI455X (gfx1250) — compile-verified, matrix operand order checked by reading, not run on hardware
//
#include <hip/hip_runtime.h>

typedef __attribute__((ext_vector_type(16))) _Float16 v16h;
typedef __attribute__((ext_vector_type(8)))  float    v8f;
typedef __attribute__((ext_vector_type(4)))  int      v4i;

#define TM 64
#define TN 64
#define TK 32
#define NTHREADS 128

// ---- optional gfx1250 async global->LDS path (guarded; falls back to plain copy) ----
#if defined(__HIP_DEVICE_COMPILE__) && __has_builtin(__builtin_amdgcn_global_load_async_to_lds_b128)
#define USE_ASYNC_A 1
#else
#define USE_ASYNC_A 0
#endif

#if USE_ASYNC_A
typedef __attribute__((address_space(1))) v4i g1_v4i;
typedef __attribute__((address_space(3))) v4i l3_v4i;
#if __has_builtin(__builtin_amdgcn_s_wait_asynccnt)
#define ASYNC_WAIT() __builtin_amdgcn_s_wait_asynccnt(0)
#else
#define ASYNC_WAIT() asm volatile("s_wait_asynccnt 0" ::: "memory")
#endif
#else
#define ASYNC_WAIT() ((void)0)
#endif

// Pack two f32 -> two f16 in one dword.
__device__ __forceinline__ unsigned int pack2h(float a, float b) {
    union { _Float16 h[2]; unsigned int u; } p;
    p.h[0] = (_Float16)a;
    p.h[1] = (_Float16)b;
    return p.u;
}

// 16-bit WMMA fragment layout. Forward: half i on lane (kg*16+m) holds
// k = i + 8*kg + ((i>=8)?8:0).  Inverse: kg=(k>>3)&1, i=(k&7)+((k>>4)<<3).
__device__ __forceinline__ int frag_kg(int k) { return (k >> 3) & 1; }
__device__ __forceinline__ int frag_i(int k)  { return (k & 7) + ((k >> 4) << 3); }

// Pre-pass: repack f32 OIHW weights into f16 tiles in fragment-swizzled order.
// Tile (tm, tk) = 2048 halves at out[(tm*KP + tk)*2048]; within a tile the
// layout equals the LDS A-tile layout, so the conv kernel A-stage is a memcpy.
__global__ void repack_weights_f16(const float* __restrict__ W, _Float16* __restrict__ out,
                                   int Cout, int KK, int KP, int total)
{
    int idx = blockIdx.x * 256 + threadIdx.x;
    if (idx >= total) return;
    int tile = idx >> 11;
    int rem  = idx & 2047;
    int tm   = tile / KP;
    int tk   = tile - tm * KP;
    int i    = rem & 15;
    int lf   = rem >> 4;
    int mloc = lf & 15;
    int kg   = (lf >> 4) & 1;
    int strip= lf >> 5;
    int row  = strip * 16 + mloc;
    int k    = i + 8 * kg + ((i >= 8) ? 8 : 0);
    int co   = tm * 64 + row;
    int gk   = tk * 32 + k;
    float v  = 0.0f;
    if (co < Cout && gk < KK) v = W[(size_t)co * KK + gk];
    out[idx] = (_Float16)v;
}

// Implicit-GEMM conv, CDNA5 WMMA, wave32. M=Cout, N=B*Ho*Wo, K=Cin*KS*KS.
// 128 threads = 4 waves, 64x64 macro tile, K stepped by 32.
// Double-buffered LDS, one barrier per K-step; A tile async-copied (if available)
// from pre-swizzled f16 weights; B tile im2col-gathered with KS a template constant.
template <int KS>
__global__ __launch_bounds__(NTHREADS) void conv_wmma_t(
    const float* __restrict__ X, const _Float16* __restrict__ Wr,
    const float* __restrict__ Bias, float* __restrict__ Y,
    int B, int Cin, int H, int W, int Cout, int Ho, int Wo,
    int stride, int pad, int relu, int out_mode, int perBatch, int scaleOff,
    int KP, int KK)
{
    constexpr int KS2 = KS * KS;
    __shared__ __align__(16) uint4 ldsA[2][256];   // 2 x 4 KB (64x32 f16)
    __shared__ __align__(16) uint4 ldsB[2][512];   // 2 x 8 KB (32x64 f16)

    const int HoWo = Ho * Wo;
    const int Ntot = B * HoWo;
    const int HW   = H * W;

    const int tid  = threadIdx.x;
    const int wave = tid >> 5;
    const int lane = tid & 31;
    const int mloc = lane & 15;
    const int kg   = lane >> 4;
    const int m0   = blockIdx.x * TM;
    const int n0   = blockIdx.y * TN;

    // ---- loop-invariant B-staging coordinates (128 % 64 == 0 -> nl fixed) ----
    const int nl  = tid & 63;
    const int kpb = tid >> 6;          // 0 or 1
    const int n   = n0 + nl;
    const bool nvalid = n < Ntot;
    int ihb = 0, iwb = 0;
    size_t nbase = 0;
    if (nvalid) {
        int bN = n / HoWo;
        int pix = n - bN * HoWo;
        int ohN = pix / Wo;
        int owN = pix - ohN * Wo;
        ihb = ohN * stride - pad;
        iwb = owN * stride - pad;
        nbase = (size_t)bN * Cin * HW;
    }

    auto stageA = [&](int ts, int buf) {
        const uint4* src = (const uint4*)Wr + (((size_t)blockIdx.x * KP + ts) << 8);
        uint4* dst = ldsA[buf];
#if USE_ASYNC_A
        __builtin_amdgcn_global_load_async_to_lds_b128(
            (g1_v4i*)(uintptr_t)(src + tid),
            (l3_v4i*)(uintptr_t)(dst + tid), 0, 0);
        __builtin_amdgcn_global_load_async_to_lds_b128(
            (g1_v4i*)(uintptr_t)(src + tid + 128),
            (l3_v4i*)(uintptr_t)(dst + tid + 128), 0, 0);
#else
        dst[tid]       = src[tid];
        dst[tid + 128] = src[tid + 128];
#endif
    };

    auto stageB = [&](int k0, int buf) {
        unsigned int* d = reinterpret_cast<unsigned int*>(ldsB[buf]);
        #pragma unroll
        for (int it = 0; it < 8; ++it) {
            int kp = kpb + it * 2;
            int k  = kp * 2;
            float v0 = 0.0f, v1 = 0.0f;
            if (nvalid) {
                int gk = k0 + k;
                if (gk < KK) {
                    int cin = gk / KS2;
                    int rem = gk - cin * KS2;
                    int kh  = rem / KS;
                    int kw  = rem - kh * KS;
                    int ih  = ihb + kh, iw = iwb + kw;
                    if ((unsigned)ih < (unsigned)H && (unsigned)iw < (unsigned)W)
                        v0 = X[nbase + (size_t)cin * HW + ih * W + iw];
                }
                int gk1 = gk + 1;
                if (gk1 < KK) {
                    int cin = gk1 / KS2;
                    int rem = gk1 - cin * KS2;
                    int kh  = rem / KS;
                    int kw  = rem - kh * KS;
                    int ih  = ihb + kh, iw = iwb + kw;
                    if ((unsigned)ih < (unsigned)H && (unsigned)iw < (unsigned)W)
                        v1 = X[nbase + (size_t)cin * HW + ih * W + iw];
                }
            }
            int hb = ((nl >> 4) * 32 + frag_kg(k) * 16 + (nl & 15)) * 16 + frag_i(k);
            d[hb >> 1] = pack2h(v0, v1);
        }
    };

    union AccU { v8f v; float f[8]; };
    AccU acc[4];
    {
        v8f z = {};
        #pragma unroll
        for (int j = 0; j < 4; ++j) acc[j].v = z;
    }

    // ---- software pipeline: stage 0, then overlap stage(s+1) with wmma(s) ----
    stageA(0, 0);
    stageB(0, 0);
    ASYNC_WAIT();
    __syncthreads();

    for (int s = 0; s < KP; ++s) {
        int cur = s & 1;
        if (s + 1 < KP) {
            stageA(s + 1, cur ^ 1);
            stageB((s + 1) * TK, cur ^ 1);
        }
        const _Float16* Ah = reinterpret_cast<const _Float16*>(ldsA[cur]);
        const _Float16* Bh = reinterpret_cast<const _Float16*>(ldsB[cur]);
        v16h af = *reinterpret_cast<const v16h*>(Ah + (wave * 32 + lane) * 16);
        #pragma unroll
        for (int j = 0; j < 4; ++j) {
            v16h bf = *reinterpret_cast<const v16h*>(Bh + (j * 32 + lane) * 16);
            acc[j].v = __builtin_amdgcn_wmma_f32_16x16x32_f16(
                false, af, false, bf, (short)0, acc[j].v, false, false);
        }
        ASYNC_WAIT();
        __syncthreads();
    }

    // ---- epilogue: C/D layout VGPR r -> row (r + 8*kg), col = mloc ----
    #pragma unroll
    for (int j = 0; j < 4; ++j) {
        int nn = n0 + j * 16 + mloc;
        if (nn >= Ntot) continue;
        int b   = nn / HoWo;
        int pix = nn - b * HoWo;
        int oh  = pix / Wo;
        int ow  = pix - oh * Wo;
        #pragma unroll
        for (int r = 0; r < 8; ++r) {
            int m  = r + 8 * kg;
            int co = m0 + wave * 16 + m;
            if (co >= Cout) continue;
            float v = acc[j].f[r] + Bias[co];
            if (relu) v = fmaxf(v, 0.0f);
            if (out_mode == 0)
                Y[(((size_t)b * Cout + co) * Ho + oh) * Wo + ow] = v;
            else
                Y[(size_t)b * perBatch + scaleOff + (oh * Wo + ow) * Cout + co] = v;
        }
    }
}

__global__ void copy_f32(const float* __restrict__ src, float* __restrict__ dst, int n)
{
    int i = blockIdx.x * blockDim.x + threadIdx.x;
    if (i < n) dst[i] = src[i];
}

static inline int conv_out_dim(int H, int ks, int s, int p) {
    return (H + 2 * p - ks) / s + 1;
}

// Launch repack + conv for one layer.
static void run_conv(hipStream_t st, const float* X, const void* Wf, const void* Bi,
                     float* Y, _Float16* Wr,
                     int B, int Cin, int H, int W, int Cout, int ks, int s, int p,
                     int relu, int out_mode, int perBatch, int scaleOff)
{
    int KK = Cin * ks * ks;
    int KP = (KK + TK - 1) / TK;
    int MP = (Cout + TM - 1) / TM;
    int total = MP * KP * 2048;
    repack_weights_f16<<<(total + 255) / 256, 256, 0, st>>>(
        (const float*)Wf, Wr, Cout, KK, KP, total);

    int Ho = conv_out_dim(H, ks, s, p);
    int Wo = conv_out_dim(W, ks, s, p);
    int Ntot = B * Ho * Wo;
    dim3 grid(MP, (Ntot + TN - 1) / TN, 1);
    if (ks == 1)
        conv_wmma_t<1><<<grid, NTHREADS, 0, st>>>(
            X, Wr, (const float*)Bi, Y, B, Cin, H, W, Cout, Ho, Wo,
            s, p, relu, out_mode, perBatch, scaleOff, KP, KK);
    else
        conv_wmma_t<3><<<grid, NTHREADS, 0, st>>>(
            X, Wr, (const float*)Bi, Y, B, Cin, H, W, Cout, Ho, Wo,
            s, p, relu, out_mode, perBatch, scaleOff, KP, KK);
}

extern "C" void kernel_launch(void* const* d_in, const int* in_sizes, int n_in,
                              void* d_out, int out_size, void* d_ws, size_t ws_size,
                              hipStream_t stream)
{
    (void)in_sizes; (void)n_in; (void)out_size; (void)ws_size;

    const float* x = (const float*)d_in[0];
    float* out = (float*)d_out;
    const int B = 8;

    // ---- byte-based workspace carver (256B aligned) ----
    char* base = (char*)d_ws;
    size_t off = 0;
    auto carveF = [&](size_t nelem) {                 // f32 activations
        float* p = (float*)(base + off);
        off = (off + nelem * sizeof(float) + 255) & ~(size_t)255;
        return p;
    };
    auto carveH = [&](int Cout, int Cin, int ks) {    // f16 repacked weights
        int KK = Cin * ks * ks;
        size_t nh = (size_t)((Cout + TM - 1) / TM) * ((KK + TK - 1) / TK) * 2048;
        _Float16* p = (_Float16*)(base + off);
        off = (off + nh * sizeof(_Float16) + 255) & ~(size_t)255;
        return p;
    };

    float* a1 = carveF((size_t)B * 64   * 150 * 150);
    float* a2 = carveF((size_t)B * 128  * 75  * 75);
    float* a3 = carveF((size_t)B * 256  * 38  * 38);
    float* a4 = carveF((size_t)B * 512  * 38  * 38);   // source 0
    float* a5 = carveF((size_t)B * 1024 * 19  * 19);   // source 1
    float* e1 = carveF((size_t)B * 256  * 19  * 19);
    float* e2 = carveF((size_t)B * 512  * 10  * 10);   // source 2
    float* e3 = carveF((size_t)B * 128  * 10  * 10);
    float* e4 = carveF((size_t)B * 256  * 5   * 5);    // source 3
    float* e5 = carveF((size_t)B * 128  * 5   * 5);
    float* e6 = carveF((size_t)B * 256  * 3   * 3);    // source 4
    float* e7 = carveF((size_t)B * 128  * 3   * 3);
    float* e8 = carveF((size_t)B * 256  * 1   * 1);    // source 5

    // ---- base network ----
    struct Spec { int ci, co, ks, s, p; };
    const Spec bs[5] = {{3,64,3,2,1},{64,128,3,2,1},{128,256,3,2,1},{256,512,3,1,1},{512,1024,3,2,1}};
    const Spec es[8] = {{1024,256,1,1,0},{256,512,3,2,1},{512,128,1,1,0},{128,256,3,2,1},
                        {256,128,1,1,0},{128,256,3,1,0},{256,128,1,1,0},{128,256,3,1,0}};
    _Float16* wrB[5]; for (int i = 0; i < 5; ++i) wrB[i] = carveH(bs[i].co, bs[i].ci, bs[i].ks);
    _Float16* wrE[8]; for (int i = 0; i < 8; ++i) wrE[i] = carveH(es[i].co, es[i].ci, es[i].ks);

    const int sch[6]  = {512, 1024, 512, 256, 256, 256};
    const int sHW[6]  = {38, 19, 10, 5, 3, 1};
    const int mbox[6] = {4, 6, 6, 6, 4, 4};
    const int NUM_CLASSES = 21;
    _Float16* wrL[6]; for (int i = 0; i < 6; ++i) wrL[i] = carveH(mbox[i] * 4,           sch[i], 3);
    _Float16* wrC[6]; for (int i = 0; i < 6; ++i) wrC[i] = carveH(mbox[i] * NUM_CLASSES, sch[i], 3);

    run_conv(stream, x,  d_in[1], d_in[2],  a1, wrB[0], B, 3,    300, 300, 64,   3, 2, 1, 1, 0, 0, 0);
    run_conv(stream, a1, d_in[3], d_in[4],  a2, wrB[1], B, 64,   150, 150, 128,  3, 2, 1, 1, 0, 0, 0);
    run_conv(stream, a2, d_in[5], d_in[6],  a3, wrB[2], B, 128,  75,  75,  256,  3, 2, 1, 1, 0, 0, 0);
    run_conv(stream, a3, d_in[7], d_in[8],  a4, wrB[3], B, 256,  38,  38,  512,  3, 1, 1, 1, 0, 0, 0);
    run_conv(stream, a4, d_in[9], d_in[10], a5, wrB[4], B, 512,  38,  38,  1024, 3, 2, 1, 1, 0, 0, 0);

    float* ebuf[9] = {a5, e1, e2, e3, e4, e5, e6, e7, e8};
    const int eH[8] = {19, 19, 10, 10, 5, 5, 3, 3};
    for (int i = 0; i < 8; ++i)
        run_conv(stream, ebuf[i], d_in[11 + 2 * i], d_in[12 + 2 * i], ebuf[i + 1], wrE[i],
                 B, es[i].ci, eH[i], eH[i], es[i].co, es[i].ks, es[i].s, es[i].p, 1, 0, 0, 0);

    // ---- heads: NHWC-flattened concat directly into d_out ----
    const float* srcs[6] = {a4, a5, e2, e4, e6, e8};
    const int locPB  = 8732 * 4;
    const int confPB = 8732 * NUM_CLASSES;
    float* locOut    = out;
    float* confOut   = out + (size_t)B * locPB;
    float* priorsOut = out + (size_t)B * (locPB + confPB);

    int locOff = 0, confOff = 0;
    for (int k = 0; k < 6; ++k) {
        int hw = sHW[k];
        run_conv(stream, srcs[k], d_in[27 + 2 * k], d_in[28 + 2 * k], locOut, wrL[k],
                 B, sch[k], hw, hw, mbox[k] * 4, 3, 1, 1, 0, 1, locPB, locOff);
        run_conv(stream, srcs[k], d_in[39 + 2 * k], d_in[40 + 2 * k], confOut, wrC[k],
                 B, sch[k], hw, hw, mbox[k] * NUM_CLASSES, 3, 1, 1, 0, 1, confPB, confOff);
        locOff  += hw * hw * mbox[k] * 4;
        confOff += hw * hw * mbox[k] * NUM_CLASSES;
    }

    // ---- priors pass-through (input 51) ----
    int np = 8732 * 4;
    copy_f32<<<(np + 255) / 256, 256, 0, stream>>>((const float*)d_in[51], priorsOut, np);
}
